// DyResConv_54571854463480
// MI455X (gfx1250) — matrix operand order repaired and checked by reading, not compiled, not run on hardware
//
#include <hip/hip_runtime.h>
#include <hip/hip_bf16.h>
#include <math.h>

// ---------------- problem constants ----------------
#define NB 32
#define NC 256
#define NH 60
#define NW 60
#define NO 256
#define NE 3
#define NSQ 16
#define NPIX (NH * NW)          // 3600
#define MTILES (NPIX / 16)      // 225
#define PH (NH + 2)             // 62 (padded)
#define PW (NW + 2)             // 62
#define PPIX (PH * PW)          // 3844
#define BM 48                   // block M tile (pixels)
#define MBLK (NPIX / BM)        // 75 blocks per sample
#define LDA 264                 // padded LDS row (256 + 8) -> conflict-free b128 reads

typedef __attribute__((ext_vector_type(16))) __bf16 v16bf;
typedef __attribute__((ext_vector_type(8)))  __bf16 v8bf;
typedef __attribute__((ext_vector_type(8)))  float  v8f;

// ---------------- kernel 1: adaptive pools + bicubic -> att (b, 3C, 5,5) ----------------
__device__ __forceinline__ float cubicW(float t) {
    const float a = -0.75f;
    t = fabsf(t);
    if (t <= 1.f) return ((a + 2.f) * t - (a + 3.f)) * t * t + 1.f;
    if (t <  2.f) return (((t - 5.f) * t + 8.f) * t - 4.f) * a;
    return 0.f;
}

__global__ __launch_bounds__(256) void pool_kernel(const float* __restrict__ x,
                                                   float* __restrict__ att) {
    const int bc = blockIdx.x;            // b*256 + c
    const int b  = bc >> 8;
    const int c  = bc & 255;
    __shared__ float s1;
    __shared__ float s3[9];
    __shared__ float s5[25];
    const int tid = threadIdx.x;
    if (tid == 0)  s1 = 0.f;
    if (tid < 9)   s3[tid] = 0.f;
    if (tid < 25)  s5[tid] = 0.f;
    __syncthreads();

    const float* xp = x + (size_t)bc * NPIX;
    float l1 = 0.f;
    for (int p = tid; p < NPIX; p += 256) {
        const float v = xp[p];
        const int row = p / NW, col = p % NW;
        l1 += v;
        atomicAdd(&s3[(row / 20) * 3 + col / 20], v);   // 3x3 pool, 20x20 regions
        atomicAdd(&s5[(row / 12) * 5 + col / 12], v);   // 5x5 pool, 12x12 regions
    }
    atomicAdd(&s1, l1);
    __syncthreads();

    if (tid < 25) {
        // bicubic 3->5 interpolation matrix (border-replicated 4-tap, a=-0.75)
        float M[5][3];
#pragma unroll
        for (int o = 0; o < 5; ++o) {
            M[o][0] = M[o][1] = M[o][2] = 0.f;
            const float xx = (o + 0.5f) * 0.6f - 0.5f;
            const float x0 = floorf(xx);
            const float f  = xx - x0;
            const int  ix0 = (int)x0;
#pragma unroll
            for (int j = -1; j <= 2; ++j) {
                int idx = ix0 + j;
                idx = idx < 0 ? 0 : (idx > 2 ? 2 : idx);
                M[o][idx] += cubicW(f - (float)j);
            }
        }
        const int o = tid / 5, q = tid % 5;
        float a3 = 0.f;
#pragma unroll
        for (int i = 0; i < 3; ++i)
#pragma unroll
            for (int j = 0; j < 3; ++j)
                a3 += M[o][i] * M[q][j] * s3[i * 3 + j] * (1.f / 400.f);

        float* attb = att + (size_t)b * (3 * NC) * 25;
        attb[(size_t)c * 25 + tid]            = s1 * (1.f / 3600.f);
        attb[(size_t)(NC + c) * 25 + tid]     = a3;
        attb[(size_t)(2 * NC + c) * 25 + tid] = s5[tid] * (1.f / 144.f);
    }
}

// ---------------- kernel 2: squeeze convs + sigmoid -> r (b, 3, 256) ----------------
__global__ __launch_bounds__(256) void routing_kernel(const float* __restrict__ att,
                                                      const float* __restrict__ rw1,
                                                      const float* __restrict__ rdw1,
                                                      const float* __restrict__ rdw2,
                                                      const float* __restrict__ rw2,
                                                      float* __restrict__ rbuf) {
    const int b   = blockIdx.x;
    const int tid = threadIdx.x;
    __shared__ float h1[NSQ][25];
    __shared__ float h2[NSQ][9];
    __shared__ float h3[NSQ];
    const float* attb = att + (size_t)b * (3 * NC) * 25;

    for (int idx = tid; idx < NSQ * 25; idx += 256) {       // 1x1: 3C -> 16
        const int s = idx / 25, p = idx % 25;
        const float* w = rw1 + (size_t)s * (3 * NC);
        float acc = 0.f;
        for (int c3 = 0; c3 < 3 * NC; ++c3) acc += w[c3] * attb[(size_t)c3 * 25 + p];
        h1[s][p] = fmaxf(acc, 0.f);
    }
    __syncthreads();
    for (int idx = tid; idx < NSQ * 9; idx += 256) {        // dw 3x3 (5->3)
        const int s = idx / 9, q = idx % 9, i = q / 3, j = q % 3;
        float acc = 0.f;
#pragma unroll
        for (int u = 0; u < 3; ++u)
#pragma unroll
            for (int v = 0; v < 3; ++v)
                acc += h1[s][(i + u) * 5 + (j + v)] * rdw1[s * 9 + u * 3 + v];
        h2[s][q] = fmaxf(acc, 0.f);
    }
    __syncthreads();
    if (tid < NSQ) {                                        // dw 3x3 (3->1)
        float acc = 0.f;
#pragma unroll
        for (int q = 0; q < 9; ++q) acc += h2[tid][q] * rdw2[tid * 9 + q];
        h3[tid] = fmaxf(acc, 0.f);
    }
    __syncthreads();
    for (int ec = tid; ec < NE * NC; ec += 256) {           // 1x1: 16 -> 768 + sigmoid
        float acc = 0.f;
#pragma unroll
        for (int s = 0; s < NSQ; ++s) acc += rw2[ec * NSQ + s] * h3[s];
        rbuf[(size_t)b * (NE * NC) + ec] = 1.f / (1.f + expf(-acc));
    }
}

// ---------------- kernel 3: combine experts -> wT[b, kpos, o, c] bf16 ----------------
__global__ __launch_bounds__(256) void wcomb_kernel(const float* __restrict__ rbuf,
                                                    const float* __restrict__ wexp,
                                                    __bf16* __restrict__ wT) {
    const int bo = blockIdx.x;           // b*256 + o
    const int b  = bo >> 8;
    const int o  = bo & 255;
    const int c  = threadIdx.x;
    const float r0 = rbuf[(size_t)b * 768 + c];
    const float r1 = rbuf[(size_t)b * 768 + 256 + c];
    const float r2 = rbuf[(size_t)b * 768 + 512 + c];
    const float* w0 = wexp + ((size_t)(0 * NO + o) * NC + c) * 9;
    const float* w1 = wexp + ((size_t)(1 * NO + o) * NC + c) * 9;
    const float* w2 = wexp + ((size_t)(2 * NO + o) * NC + c) * 9;
#pragma unroll
    for (int kpos = 0; kpos < 9; ++kpos) {
        const float acc = r0 * w0[kpos] + r1 * w1[kpos] + r2 * w2[kpos];
        wT[(((size_t)b * 9 + kpos) * NO + o) * NC + c] = (__bf16)acc;
    }
}

// ---------------- kernel 4a: zero the 1-pixel halo of padded N(HW)C activations ------
__global__ __launch_bounds__(256) void halo_kernel(__bf16* __restrict__ xTpad) {
    const int b = blockIdx.x;
    const int c = threadIdx.x;
    __bf16* base = xTpad + (size_t)b * PPIX * NC;
    for (int i = 0; i < 2 * PW + 2 * NH; ++i) {   // 244 halo pixels
        int ph, pw;
        if      (i < PW)          { ph = 0;               pw = i; }
        else if (i < 2 * PW)      { ph = PH - 1;          pw = i - PW; }
        else if (i < 2 * PW + NH) { ph = 1 + (i - 2*PW);  pw = 0; }
        else                      { ph = 1 + (i - 2*PW - NH); pw = PW - 1; }
        base[(size_t)(ph * PW + pw) * NC + c] = (__bf16)0.f;
    }
}

// ---------------- kernel 4b: x NCHW -> padded N(HW)C bf16 (LDS tiled transpose) ------
__global__ __launch_bounds__(256) void xT_kernel(const float* __restrict__ x,
                                                 __bf16* __restrict__ xTpad) {
    const int b     = blockIdx.x / MTILES;
    const int pt    = blockIdx.x % MTILES;
    const int pbase = pt * 16;
    __shared__ float sm[16][17];          // [c][p], padded against bank conflicts
    const int tid = threadIdx.x;
    const int cl = tid >> 4, pl = tid & 15;
    const int pOut = pbase + (tid >> 4);  // flat unpadded pixel this thread writes
    const int oh = pOut / NW, ow = pOut % NW;
    const size_t padRow = (size_t)b * PPIX + (oh + 1) * PW + (ow + 1);
    for (int cbase = 0; cbase < NC; cbase += 16) {
        sm[cl][pl] = x[((size_t)b * NC + cbase + cl) * NPIX + pbase + pl];
        __syncthreads();
        xTpad[padRow * NC + cbase + (tid & 15)] = (__bf16)sm[tid & 15][tid >> 4];
        __syncthreads();
    }
}

// ---------------- kernel 5: implicit-GEMM conv via v_wmma_f32_16x16x32_bf16 ----------
// Block: 8 waves, tile 48(M=pixels) x 256(N). Wave: 48x32 = 6 accumulators.
// A rows (48 pix x 256 ch per kernel position, 24 KB) staged in LDS, double-buffered
// across the 9 kernel positions; B fragments streamed from global (channel-contig).
// K-loop: 9 kpos x 8 chunks x 6 WMMAs = 432 WMMAs per wave.
__global__ __launch_bounds__(256) void gemm_kernel(const __bf16* __restrict__ xTpad,
                                                   const __bf16* __restrict__ wT,
                                                   float* __restrict__ y) {
    const int mblk  = blockIdx.x % MBLK;
    const int b     = blockIdx.x / MBLK;
    const int tid   = threadIdx.x;
    const int lane  = tid & 31;
    const int wave  = tid >> 5;
    const int hiHalf = lane >> 4;
    const int mbase  = mblk * BM;
    const int nbase  = wave * 32;
    const int nIdx0  = nbase + (lane & 15);    // first B column / output channel

    __shared__ __bf16 As[2][BM][LDA];          // double-buffered A tile, padded rows

    const __bf16* xb = xTpad + (size_t)b * PPIX * NC;
    const __bf16* wb = wT + (size_t)b * 9 * NO * NC;

    // Cooperative A staging plan: thread t copies 6 x 16B chunks.
    // pixel = (t>>5) + i*8 (i=0..5), channel group = (t&31)*8.
    const int cg = (lane) * 8 + (wave & 0) * 0;          // (t&31)*8
    const int cgOff = (tid & 31) * 8;
    int bp[6];                                            // padded row base per pixel
#pragma unroll
    for (int i = 0; i < 6; ++i) {
        const int pix = (tid >> 5) + i * 8;
        const int p   = mbase + pix;
        const int oh  = p / NW, ow = p % NW;
        bp[i] = (oh + 1) * PW + (ow + 1);
    }
    (void)cg;

    // preload kernel position 0 (kh=kw=0 -> offset -PW-1)
    {
        v8bf st[6];
#pragma unroll
        for (int i = 0; i < 6; ++i)
            st[i] = *(const v8bf*)(xb + (size_t)(bp[i] - PW - 1) * NC + cgOff);
#pragma unroll
        for (int i = 0; i < 6; ++i)
            *(v8bf*)(&As[0][(tid >> 5) + i * 8][cgOff]) = st[i];
    }
    __syncthreads();

    v8f acc[3][2];
#pragma unroll
    for (int ms = 0; ms < 3; ++ms) { acc[ms][0] = (v8f){}; acc[ms][1] = (v8f){}; }

    for (int kpos = 0; kpos < 9; ++kpos) {
        const int cur = kpos & 1;
        // issue global loads for next kernel position early (hide behind WMMAs)
        v8bf st[6];
        if (kpos < 8) {
            const int kh = (kpos + 1) / 3, kw = (kpos + 1) % 3;
            const int roff = (kh - 1) * PW + (kw - 1);
#pragma unroll
            for (int i = 0; i < 6; ++i)
                st[i] = *(const v8bf*)(xb + (size_t)(bp[i] + roff) * NC + cgOff);
            __builtin_prefetch((const void*)(wb + ((size_t)(kpos + 1) * NO + nIdx0) * NC), 0, 1);
        }

        const __bf16* brow0 = wb + ((size_t)kpos * NO + nIdx0) * NC;
        const __bf16* brow1 = brow0 + 16 * NC;

#pragma unroll
        for (int cb = 0; cb < NC; cb += 32) {
            // B fragments, 32x16 16-bit layout: lanes 0-15 K {0..7,16..23}; lanes 16-31 K {8..15,24..31}
            const v8bf b00 = *(const v8bf*)(brow0 + cb + hiHalf * 8);
            const v8bf b01 = *(const v8bf*)(brow0 + cb + hiHalf * 8 + 16);
            const v8bf b10 = *(const v8bf*)(brow1 + cb + hiHalf * 8);
            const v8bf b11 = *(const v8bf*)(brow1 + cb + hiHalf * 8 + 16);
            v16bf bm0, bm1;
#pragma unroll
            for (int i = 0; i < 8; ++i) {
                bm0[i] = b00[i]; bm0[8 + i] = b01[i];
                bm1[i] = b10[i]; bm1[8 + i] = b11[i];
            }
#pragma unroll
            for (int ms = 0; ms < 3; ++ms) {
                // A fragment from LDS, 16-bit A 16x32 layout:
                // lanes 0-15 K {0..7,16..23}; lanes 16-31 K {8..15,24..31}
                const __bf16* ar = &As[cur][ms * 16 + (lane & 15)][0];
                const v8bf a0 = *(const v8bf*)(ar + cb + hiHalf * 8);
                const v8bf a1 = *(const v8bf*)(ar + cb + hiHalf * 8 + 16);
                v16bf a;
#pragma unroll
                for (int i = 0; i < 8; ++i) { a[i] = a0[i]; a[8 + i] = a1[i]; }
                acc[ms][0] = __builtin_amdgcn_wmma_f32_16x16x32_bf16(
                    false, a, false, bm0, (short)0, acc[ms][0], false, false);
                acc[ms][1] = __builtin_amdgcn_wmma_f32_16x16x32_bf16(
                    false, a, false, bm1, (short)0, acc[ms][1], false, false);
            }
        }

        if (kpos < 8) {
#pragma unroll
            for (int i = 0; i < 6; ++i)
                *(v8bf*)(&As[cur ^ 1][(tid >> 5) + i * 8][cgOff]) = st[i];
        }
        __syncthreads();
    }

    // C/D layout: VGPR r <-> M = r + 8*hiHalf, N = lane&15 => contiguous pixels per half
#pragma unroll
    for (int ms = 0; ms < 3; ++ms) {
#pragma unroll
        for (int ns = 0; ns < 2; ++ns) {
            float* yb = y + ((size_t)b * NO + nIdx0 + ns * 16) * NPIX
                          + mbase + ms * 16 + hiHalf * 8;
            const v8f v = acc[ms][ns];
            *(float4*)(yb)     = make_float4(v[0], v[1], v[2], v[3]);
            *(float4*)(yb + 4) = make_float4(v[4], v[5], v[6], v[7]);
        }
    }
}

// ---------------- launcher ----------------
extern "C" void kernel_launch(void* const* d_in, const int* in_sizes, int n_in,
                              void* d_out, int out_size, void* d_ws, size_t ws_size,
                              hipStream_t stream) {
    const float* x     = (const float*)d_in[0];   // (32,256,60,60)
    const float* wexp  = (const float*)d_in[1];   // (3,256,256,3,3)
    const float* rw1   = (const float*)d_in[2];   // (16,768,1,1)
    const float* rdw1  = (const float*)d_in[3];   // (16,1,3,3)
    const float* rdw2  = (const float*)d_in[4];   // (16,1,3,3)
    const float* rw2   = (const float*)d_in[5];   // (768,16,1,1)
    float* y = (float*)d_out;                     // (32,256,60,60) f32

    // workspace layout (256B aligned)
    char* ws = (char*)d_ws;
    float*  att   = (float*)(ws);                 // 32*768*25*4     =  2,457,600 B
    float*  rbuf  = (float*)(ws + 2457600);       // 32*768*4        =     98,304 B
    __bf16* xTpad = (__bf16*)(ws + 2555904);      // 32*3844*256*2   = 62,980,096 B
    __bf16* wT    = (__bf16*)(ws + 65536000);     // 32*9*256*256*2  = 37,748,736 B
                                                  // total           ~103.3 MB

    pool_kernel   <<<NB * NC,     256, 0, stream>>>(x, att);
    routing_kernel<<<NB,          256, 0, stream>>>(att, rw1, rdw1, rdw2, rw2, rbuf);
    wcomb_kernel  <<<NB * NO,     256, 0, stream>>>(rbuf, wexp, wT);
    halo_kernel   <<<NB,          256, 0, stream>>>(xTpad);
    xT_kernel     <<<NB * MTILES, 256, 0, stream>>>(x, xTpad);
    gemm_kernel   <<<NB * MBLK,   256, 0, stream>>>(xTpad, wT, y);
}
